// BiLSTM_CRF_7868380086627
// MI455X (gfx1250) — compile-verified, matrix operand order checked by reading, not run on hardware
//
#include <hip/hip_runtime.h>
#include <hip/hip_bf16.h>

// Problem dims (reference): B=32, S=512, V=50000, E=300, HID=512, H=256, T=16
// Only batch row 0 reaches the output -> compute b=0 only (32x work cut).
#define Bv     32
#define Sv     512
#define Ev     300
#define EP     320      // E padded to a multiple of 32 (WMMA K-step)
#define HIDv   512
#define Hv     256
#define G4     1024     // 4*H
#define Tt     16
#define STARTt 14
#define STOPt  15
#define NEGv   (-10000.0f)

typedef __attribute__((ext_vector_type(16))) __bf16    v16bf;
typedef __attribute__((ext_vector_type(8)))  float     v8f;
typedef __attribute__((ext_vector_type(4)))  unsigned  v4u;
typedef __attribute__((ext_vector_type(8)))  int       v8i;
typedef __attribute__((ext_vector_type(4)))  int       v4i;

__device__ __forceinline__ unsigned short f2bf(float f) {
  union { float f; unsigned u; } x; x.f = f;
  unsigned r = (x.u + 0x7fffu + ((x.u >> 16) & 1u)) >> 16;   // RNE
  return (unsigned short)r;
}
__device__ __forceinline__ __bf16 us2bf(unsigned short u) {
  return __builtin_bit_cast(__bf16, u);
}

// 16x32 bf16 fragment load, CDNA5 ISA 7.12.2 16-bit A layout:
//   lane 0-15 -> row (lane&15); VGPR v holds K pair
//   base = (v<4 ? 2v : 16+2(v-4)) + 8*(lane>=16).
// Also valid as the B-operand loader when B is given as W[n][k] row-major
// (B column n == W row n). Contiguous per lane -> compiler merges to b128.
__device__ __forceinline__ v16bf load_frag16x32(const unsigned short* p,
                                                int row0, int col0, int ld) {
  const int lane = threadIdx.x & 31;
  const int r    = lane & 15;
  const int hh   = lane >> 4;
  const unsigned short* rowp = p + (size_t)(row0 + r) * ld + col0 + 8 * hh;
  v16bf out;
#pragma unroll
  for (int v = 0; v < 8; ++v) {
    const int kb = (v < 4) ? (2 * v) : (16 + 2 * (v - 4));
    out[2 * v]     = us2bf(rowp[kb]);
    out[2 * v + 1] = us2bf(rowp[kb + 1]);
  }
  return out;
}

__device__ __forceinline__ v8f wmma_bf16(v16bf a, v16bf b, v8f c) {
  return __builtin_amdgcn_wmma_f32_16x16x32_bf16(false, a, false, b,
                                                 (short)0, c, false, false);
}

// ---------------------------------------------------------------- prep ----
__global__ void prep_x_kernel(const int* __restrict__ sentence,
                              const float* __restrict__ emb,
                              unsigned short* __restrict__ X,
                              int* __restrict__ bar) {
  int idx = blockIdx.x * blockDim.x + threadIdx.x;
  if (idx < 2) bar[idx] = 0;                 // reset cross-WG barrier counters
  if (idx >= Sv * EP) return;
  int t = idx / EP, k = idx - t * EP;
  float v = 0.0f;
  if (k < Ev) v = emb[(size_t)sentence[t] * Ev + k];   // sentence[0][t]
  X[idx] = f2bf(v);
}

__global__ void prep_w_kernel(const float* __restrict__ Wih_f, const float* __restrict__ Wih_b,
                              const float* __restrict__ Whh_f, const float* __restrict__ Whh_b,
                              const float* __restrict__ W1,    const float* __restrict__ W2,
                              unsigned short* __restrict__ WihC, unsigned short* __restrict__ WhhC,
                              unsigned short* __restrict__ W1b,  unsigned short* __restrict__ W2b) {
  int idx = blockIdx.x * blockDim.x + threadIdx.x;
  const int N0 = 2048 * EP;       // fused [W_ih_f; W_ih_b], K padded
  const int N1 = 2 * G4 * Hv;     // W_hh both dirs
  const int N2 = HIDv * HIDv;     // W1
  const int N3 = Tt * HIDv;       // W2
  if (idx < N0) {
    int n = idx / EP, k = idx - n * EP;
    float v = 0.0f;
    if (k < Ev)
      v = (n < G4) ? Wih_f[(size_t)n * Ev + k] : Wih_b[(size_t)(n - G4) * Ev + k];
    WihC[idx] = f2bf(v);
  } else if ((idx -= N0) < N1) {
    float v = (idx < G4 * Hv) ? Whh_f[idx] : Whh_b[idx - G4 * Hv];
    WhhC[idx] = f2bf(v);
  } else if ((idx -= N1) < N2) {
    W1b[idx] = f2bf(W1[idx]);
  } else if ((idx -= N2) < N3) {
    W2b[idx] = f2bf(W2[idx]);
  }
}

// ------------------------------------------- input projection GEMM --------
// Xg[512,2048] = X[512,320] @ WihC^T + (b_ih + b_hh), both directions fused.
__global__ void gemm_xg_kernel(const unsigned short* __restrict__ X,
                               const unsigned short* __restrict__ WihC,
                               const float* __restrict__ bih_f, const float* __restrict__ bhh_f,
                               const float* __restrict__ bih_b, const float* __restrict__ bhh_b,
                               float* __restrict__ Xg) {
  const int wave = threadIdx.x >> 5;
  const int lane = threadIdx.x & 31;
  const int tile = blockIdx.x * 8 + wave;    // 32*128 = 4096 tiles
  const int mt = tile >> 7;
  const int nt = tile & 127;
  v8f c = {};
#pragma unroll
  for (int k0 = 0; k0 < EP; k0 += 32) {
    v16bf a = load_frag16x32(X,    mt * 16, k0, EP);
    v16bf b = load_frag16x32(WihC, nt * 16, k0, EP);
    c = wmma_bf16(a, b, c);
  }
  const int hh = lane >> 4;
  const int n  = nt * 16 + (lane & 15);
  float bias = (n < G4) ? (bih_f[n] + bhh_f[n]) : (bih_b[n - G4] + bhh_b[n - G4]);
#pragma unroll
  for (int r = 0; r < 8; ++r) {
    int m = mt * 16 + r + 8 * hh;            // C/D layout: VGPR r -> M=r(+8)
    Xg[(size_t)m * 2048 + n] = c[r] + bias;
  }
}

// ------------------------------------------------ recurrent LSTM ----------
// 4 workgroups: (dir, half). Each WG keeps its 512x256 bf16 W_hh slice
// (256 KB) resident in LDS for all 512 steps, filled once by a single TDM
// tensor_load_to_lds. Gate halves are exchanged through an L2 ping-pong
// buffer with one global-atomic barrier per step; c-state is maintained
// redundantly in both halves so only the gates cross WGPs.
__global__ void lstm_coop_kernel(const float* __restrict__ Xg,
                                 const unsigned short* __restrict__ WhhC,
                                 const float* __restrict__ h0, const float* __restrict__ c0,
                                 unsigned short* __restrict__ hcat,
                                 float* gstage, int* bar) {
  extern __shared__ char smem[];             // 256KB Wlds + 8KB h tile
  unsigned short* Wlds = (unsigned short*)smem;              // [512][256]
  unsigned short* h_s  = (unsigned short*)(smem + 512 * Hv * 2);  // [16][256]
  const int dir  = blockIdx.x >> 1;
  const int half = blockIdx.x & 1;
  const int tid  = threadIdx.x;              // 256 threads = 8 waves
  const int lane = tid & 31;
  const int wave = tid >> 5;

  // ---- one-shot TDM fill of the LDS-resident W_hh slice (wave 0 only) ----
  if (tid < 32) {
    unsigned long long va = (unsigned long long)(WhhC + (size_t)dir * G4 * Hv
                                                       + (size_t)half * 512 * Hv);
    v4u g0 = { 1u,                                   // count=1, user descriptor
               0u,                                   // lds_addr = 0 (Wlds base)
               (unsigned)va,                         // global_addr[31:0]
               (((unsigned)(va >> 32)) & 0x1FFFFFFu) | (2u << 30) }; // addr[56:32], type=2
    v8i g1 = { 0x00010000,                           // mask=0, data_size=1 (2B)
               0x01000000,                           // tensor_dim0 = 256
               0x02000000,                           // tensor_dim1 = 512
               0x01000000,                           // tile_dim0 = 256
               0x00000200,                           // tile_dim1 = 512
               0x00000100,                           // tensor_dim0_stride = 256
               0, 0 };
    v4i z4 = { 0, 0, 0, 0 };
    v8i z8 = { 0, 0, 0, 0, 0, 0, 0, 0 };
    __builtin_amdgcn_tensor_load_to_lds(g0, g1, z4, z4, z8, 0);
    __builtin_amdgcn_s_wait_tensorcnt(0);
  }

  // ---- init h tile (rows 1..15 pinned to zero) and per-thread c state ----
  for (int i = tid; i < 16 * Hv; i += 256)
    h_s[i] = (i < Hv) ? f2bf(h0[(size_t)dir * Bv * Hv + i]) : (unsigned short)0;
  float c_reg = c0[(size_t)dir * Bv * Hv + tid];
  __syncthreads();

  volatile float* gs = gstage;               // L2 ping-pong [2][2][1024]
  for (int s = 0; s < Sv; ++s) {
    const int torig = dir ? (Sv - 1 - s) : s;
    const float* xg = Xg + (size_t)torig * 2048 + dir * G4;
    if (s + 1 < Sv) {
      const int tnext = dir ? (Sv - 2 - s) : (s + 1);
      __builtin_prefetch(Xg + (size_t)tnext * 2048 + dir * G4 + (lane << 3), 0, 1);
    }
    v16bf afr[8];                            // A fragments from LDS h tile
#pragma unroll
    for (int kk = 0; kk < 8; ++kk)
      afr[kk] = load_frag16x32(h_s, 0, kk * 32, Hv);
    const int hh = lane >> 4, nloc16 = lane & 15;
    volatile float* gw = gs + (size_t)((s & 1) * 2 + dir) * G4;
#pragma unroll
    for (int i = 0; i < 4; ++i) {            // 4 local N-tiles per wave
      const int ntl = wave * 4 + i;          // 0..31 within this half
      const int nl  = ntl * 16 + nloc16;     // 0..511 local column
      v8f c = {};
      if (hh == 0) c[0] = xg[half * 512 + nl];   // init real (m=0) row
#pragma unroll
      for (int kk = 0; kk < 8; ++kk) {
        v16bf b = load_frag16x32(Wlds, ntl * 16, kk * 32, Hv);
        c = wmma_bf16(afr[kk], b, c);
      }
      if (hh == 0) gw[half * 512 + nl] = c[0];   // stage gate row 0 in L2
    }
    // one cross-WG barrier per step (ping-pong makes a single one safe)
    if (tid == 0) {
      __threadfence();
      atomicAdd(&bar[dir], 1);
      while (atomicAdd(&bar[dir], 0) < 2 * (s + 1)) { }
      __threadfence();
    }
    __syncthreads();
    // cell update, one hidden unit per thread; both halves replicate c/h
    float xi  = gw[tid];
    float xf  = gw[Hv + tid];
    float xg2 = gw[2 * Hv + tid];
    float xo  = gw[3 * Hv + tid];
    float ig = 1.0f / (1.0f + __expf(-xi));
    float fg = 1.0f / (1.0f + __expf(-xf));
    float gg = tanhf(xg2);
    float og = 1.0f / (1.0f + __expf(-xo));
    c_reg = fg * c_reg + ig * gg;
    float hv = og * tanhf(c_reg);
    unsigned short hb = f2bf(hv);
    h_s[tid] = hb;                                           // next step's A
    if (half == 0)
      hcat[(size_t)torig * HIDv + dir * Hv + tid] = hb;      // fused concat
    __syncthreads();
  }
}

// ------------------------------------------------------- MLP GEMMs --------
__global__ void gemm_h1_kernel(const unsigned short* __restrict__ hcatU,
                               const unsigned short* __restrict__ W1b,
                               const float* __restrict__ b1,
                               unsigned short* __restrict__ hmid) {
  const int wave = threadIdx.x >> 5, lane = threadIdx.x & 31;
  const int tile = blockIdx.x * 8 + wave;    // 32*32 = 1024 tiles
  const int mt = tile >> 5, nt = tile & 31;
  v8f c = {};
#pragma unroll 4
  for (int k0 = 0; k0 < HIDv; k0 += 32) {
    v16bf a = load_frag16x32(hcatU, mt * 16, k0, HIDv);
    v16bf b = load_frag16x32(W1b,   nt * 16, k0, HIDv);
    c = wmma_bf16(a, b, c);
  }
  const int hh = lane >> 4, n = nt * 16 + (lane & 15);
  float bias = b1[n];
#pragma unroll
  for (int r = 0; r < 8; ++r) {
    int m = mt * 16 + r + 8 * hh;
    float v = c[r] + bias;
    hmid[(size_t)m * HIDv + n] = f2bf(v > 0.0f ? v : 0.0f);   // fused relu
  }
}

__global__ void gemm_feats_kernel(const unsigned short* __restrict__ hmid,
                                  const unsigned short* __restrict__ W2b,
                                  const float* __restrict__ b2,
                                  float* __restrict__ feats) {
  const int wave = threadIdx.x >> 5, lane = threadIdx.x & 31;
  const int mt = blockIdx.x * 8 + wave;      // 32 M-tiles, single N-tile (T=16)
  v8f c = {};
#pragma unroll 4
  for (int k0 = 0; k0 < HIDv; k0 += 32) {
    v16bf a = load_frag16x32(hmid, mt * 16, k0, HIDv);
    v16bf b = load_frag16x32(W2b,  0,       k0, HIDv);
    c = wmma_bf16(a, b, c);
  }
  const int hh = lane >> 4, n = lane & 15;
  float bias = b2[n];
#pragma unroll
  for (int r = 0; r < 8; ++r) {
    int m = mt * 16 + r + 8 * hh;
    feats[(size_t)m * Tt + n] = c[r] + bias;
  }
}

// -------------------------------------------------------- Viterbi ---------
__global__ void viterbi_kernel(const float* __restrict__ feats,
                               const float* __restrict__ trans,
                               int* __restrict__ bp, float* __restrict__ out) {
  const int tid = threadIdx.x;               // one wave32, lane = next-tag
  __shared__ float fv[Tt];
  __shared__ float nfv[Tt];
  __shared__ float trS[Tt * Tt];
  for (int i = tid; i < Tt * Tt; i += 32) trS[i] = trans[i];
  if (tid < Tt) fv[tid] = (tid == STARTt) ? 0.0f : NEGv;
  __syncthreads();
  for (int s = 0; s < Sv; ++s) {
    if (tid < Tt) {
      float best = -3.4e38f; int bi = 0;
#pragma unroll
      for (int p = 0; p < Tt; ++p) {         // first-max wins (jnp.argmax)
        float sc = fv[p] + trS[tid * Tt + p];
        if (sc > best) { best = sc; bi = p; }
      }
      bp[s * Tt + tid] = bi;
      nfv[tid] = best + feats[s * Tt + tid];
    }
    __syncthreads();
    if (tid < Tt) fv[tid] = nfv[tid];
    __syncthreads();
  }
  if (tid == 0) {
    float best = -3.4e38f; int bt = 0;
    for (int t = 0; t < Tt; ++t) {
      float sc = fv[t] + trS[STOPt * Tt + t];
      if (sc > best) { best = sc; bt = t; }
    }
    out[0] = best;
    int tag = bt;
    for (int s = Sv - 1; s >= 0; --s) {      // path[s]=carry; carry=bp[s][carry]
      out[1 + s] = (float)tag;
      tag = bp[s * Tt + tag];
    }
  }
}

// --------------------------------------------------------------- launch ---
extern "C" void kernel_launch(void* const* d_in, const int* in_sizes, int n_in,
                              void* d_out, int out_size, void* d_ws, size_t ws_size,
                              hipStream_t stream) {
  (void)in_sizes; (void)n_in; (void)out_size; (void)ws_size;
  const int*   sentence = (const int*)  d_in[0];
  const float* h0    = (const float*)d_in[1];
  const float* c0    = (const float*)d_in[2];
  const float* emb   = (const float*)d_in[3];
  const float* Wih_f = (const float*)d_in[4];
  const float* Whh_f = (const float*)d_in[5];
  const float* bih_f = (const float*)d_in[6];
  const float* bhh_f = (const float*)d_in[7];
  const float* Wih_b = (const float*)d_in[8];
  const float* Whh_b = (const float*)d_in[9];
  const float* bih_b = (const float*)d_in[10];
  const float* bhh_b = (const float*)d_in[11];
  const float* W1    = (const float*)d_in[12];
  const float* b1    = (const float*)d_in[13];
  const float* W2    = (const float*)d_in[14];
  const float* b2    = (const float*)d_in[15];
  const float* trans = (const float*)d_in[16];

  char* ws = (char*)d_ws;                               // ~8.6 MB used
  unsigned short* X    = (unsigned short*)(ws + 0);         // 512*320 bf16
  unsigned short* WihC = (unsigned short*)(ws + 327680);    // 2048*320 bf16
  unsigned short* WhhC = (unsigned short*)(ws + 1638400);   // 2*1024*256 bf16
  unsigned short* W1b  = (unsigned short*)(ws + 2686976);   // 512*512 bf16
  unsigned short* W2b  = (unsigned short*)(ws + 3211264);   // 16*512 bf16
  float*          Xg   = (float*)         (ws + 3227648);   // 512*2048 f32
  unsigned short* hcat = (unsigned short*)(ws + 7421952);   // 512*512 bf16
  unsigned short* hmid = (unsigned short*)(ws + 7946240);   // 512*512 bf16
  float*          feats= (float*)         (ws + 8470528);   // 512*16 f32
  int*            bp   = (int*)           (ws + 8503296);   // 512*16 i32
  int*            bar  = (int*)           (ws + 8536064);   // 2 counters
  float*          gst  = (float*)         (ws + 8536320);   // [2][2][1024] f32
  float* out = (float*)d_out;

  const int lstmShmem = 512 * Hv * 2 + 16 * Hv * 2;     // 256KB W + 8KB h
  (void)hipFuncSetAttribute((const void*)lstm_coop_kernel,
                            hipFuncAttributeMaxDynamicSharedMemorySize, lstmShmem);

  prep_x_kernel<<<(Sv * EP + 255) / 256, 256, 0, stream>>>(sentence, emb, X, bar);
  const int prepTot = 2048 * EP + 2 * G4 * Hv + HIDv * HIDv + Tt * HIDv;
  prep_w_kernel<<<(prepTot + 255) / 256, 256, 0, stream>>>(
      Wih_f, Wih_b, Whh_f, Whh_b, W1, W2, WihC, WhhC, W1b, W2b);
  gemm_xg_kernel<<<512, 256, 0, stream>>>(X, WihC, bih_f, bhh_f, bih_b, bhh_b, Xg);
  lstm_coop_kernel<<<4, 256, lstmShmem, stream>>>(Xg, WhhC, h0, c0, hcat, gst, bar);
  gemm_h1_kernel<<<128, 256, 0, stream>>>(hcat, W1b, b1, hmid);
  gemm_feats_kernel<<<4, 256, 0, stream>>>(hmid, W2b, b2, feats);
  viterbi_kernel<<<1, 32, 0, stream>>>(feats, trans, bp, out);
}
